// WaveletTransformUnit_73083163509585
// MI455X (gfx1250) — compile-verified, matrix operand order checked by reading, not run on hardware
//
#include <hip/hip_runtime.h>

#define SQ 0.7071067811865476f

typedef __bf16 bf16;
typedef __attribute__((ext_vector_type(16))) bf16  v16bf;
typedef __attribute__((ext_vector_type(8)))  bf16  v8bf;
typedef __attribute__((ext_vector_type(8)))  float v8f;

#define CAT16(a, b) __builtin_shufflevector(a, b, 0,1,2,3,4,5,6,7,8,9,10,11,12,13,14,15)

// ---------------- weight conversion ----------------
__global__ void k_cvt_w1(const float* __restrict__ w, bf16* __restrict__ wb) {
    int i = blockIdx.x * 256 + threadIdx.x;        // 256*256
    wb[i] = (bf16)w[i];
}

// convW_w [O=256][I=512][3][3] -> wW[(o*9+tap)*512 + c]
__global__ void k_cvt_w2(const float* __restrict__ w, bf16* __restrict__ wb) {
    int tid = blockIdx.x * 256 + threadIdx.x;      // 256*9*512
    int c   = tid & 511;
    int tap = (tid >> 9) % 9;
    int o   = tid / (512 * 9);
    wb[tid] = (bf16)w[((size_t)(o * 512 + c)) * 9 + tap];
}

// ---------------- branch 0: DWT -> bf16 A matrix [4*16384, 256] ----------------
__global__ __launch_bounds__(256) void k_dwt_sub(const float* __restrict__ x,
                                                 bf16* __restrict__ A) {
    const int c   = threadIdx.x;
    const int pix = blockIdx.x;                    // (b*32+hp)*32+wp
    const int wp  = pix & 31;
    const int hp  = (pix >> 5) & 31;
    const int b   = pix >> 10;
    const float* xp = x + (((size_t)(b * 256 + c) * 64 + hp * 2) * 64 + wp * 2);
    float p00 = xp[0], p01 = xp[1], p10 = xp[64], p11 = xp[65];
    float lo0 = (p00 + p10) * SQ, lo1 = (p01 + p11) * SQ;
    float hi0 = (p00 - p10) * SQ, hi1 = (p01 - p11) * SQ;
    float cA = (lo0 + lo1) * SQ, cV = (lo0 - lo1) * SQ;
    float cH = (hi0 + hi1) * SQ, cD = (hi0 - hi1) * SQ;
    const size_t m = (size_t)pix;
    A[(0 * 16384 + m) * 256 + c] = (bf16)cA;
    A[(1 * 16384 + m) * 256 + c] = (bf16)cH;
    A[(2 * 16384 + m) * 256 + c] = (bf16)cV;
    A[(3 * 16384 + m) * 256 + c] = (bf16)cD;
}

// ---- WMMA GEMM: Y[m,o] = sum_c A[m,c]*W[o,c]  (M=65536,N=256,K=256) --------------
// Block: 8 waves x (16M x 64N) per wave; B panel (64n x 256k) staged in LDS (32 KiB).
__global__ __launch_bounds__(256) void gemm_sub(const bf16* __restrict__ A,
                                                const bf16* __restrict__ Bw,
                                                float* __restrict__ Y) {
    __shared__ __align__(16) bf16 sB[64 * 256];
    const int nBase = blockIdx.y * 64;
    {
        int t = threadIdx.x;
        int n = t >> 2;                 // 0..63
        int k = (t & 3) * 64;           // 64 bf16 each
        const bf16* src = Bw + (size_t)(nBase + n) * 256 + k;
        bf16* dst = &sB[n * 256 + k];
#pragma unroll
        for (int q = 0; q < 8; ++q)
            *(v8bf*)(dst + q * 8) = *(const v8bf*)(src + q * 8);
    }
    __syncthreads();
    const int wave   = threadIdx.x >> 5;
    const int lane   = threadIdx.x & 31;
    const int laneHi = lane >> 4;
    const int row    = lane & 15;
    const int mBase  = (blockIdx.x * 8 + wave) * 16;
    const bf16* arow = A + (size_t)(mBase + row) * 256;
    v8f acc[4] = {};
#pragma unroll
    for (int k0 = 0; k0 < 256; k0 += 32) {
        v8bf a0 = *(const v8bf*)(arow + k0 + laneHi * 8);
        v8bf a1 = *(const v8bf*)(arow + k0 + 16 + laneHi * 8);
        v16bf av = CAT16(a0, a1);
#pragma unroll
        for (int nt = 0; nt < 4; ++nt) {
            const bf16* brow = &sB[(nt * 16 + row) * 256 + k0];
            v8bf b0 = *(const v8bf*)(brow + laneHi * 8);
            v8bf b1 = *(const v8bf*)(brow + 16 + laneHi * 8);
            v16bf bv = CAT16(b0, b1);
            acc[nt] = __builtin_amdgcn_wmma_f32_16x16x32_bf16(false, av, false, bv,
                                                              (short)0, acc[nt], false, false);
        }
    }
#pragma unroll
    for (int nt = 0; nt < 4; ++nt) {
        float* yout = Y + (size_t)(mBase + laneHi * 8) * 256 + nBase + nt * 16 + row;
#pragma unroll
        for (int i = 0; i < 8; ++i) yout[(size_t)i * 256] = acc[nt][i];
    }
}

// ---------------- per-(subband,channel) batch stats ----------------
__global__ __launch_bounds__(256) void k_statsY(const float* __restrict__ Y,
                                                float* __restrict__ meanY,
                                                float* __restrict__ rvarY) {
    __shared__ float ssum[256], ssq[256];
    const int so = blockIdx.x;                     // s*256+o
    const int s = so >> 8, o = so & 255;
    const float* base = Y + (size_t)s * 16384 * 256 + o;
    float sum = 0.f, sq = 0.f;
    for (int i = threadIdx.x; i < 16384; i += 256) {
        float v = base[(size_t)i * 256];
        sum += v; sq += v * v;
    }
    ssum[threadIdx.x] = sum; ssq[threadIdx.x] = sq;
    __syncthreads();
    for (int st = 128; st > 0; st >>= 1) {
        if (threadIdx.x < st) {
            ssum[threadIdx.x] += ssum[threadIdx.x + st];
            ssq[threadIdx.x]  += ssq[threadIdx.x + st];
        }
        __syncthreads();
    }
    if (threadIdx.x == 0) {
        float m   = ssum[0] * (1.0f / 16384.f);
        float var = ssq[0] * (1.0f / 16384.f) - m * m;
        meanY[so] = m;
        rvarY[so] = rsqrtf(var + 1e-5f);
    }
}

// ---------------- BN+ReLU on 4 subbands, IDWT, write rimg0 (NHWC bf16, ch 0..255) ----
__global__ __launch_bounds__(256) void k_bn_idwt(const float* __restrict__ Y,
        const float* __restrict__ meanY, const float* __restrict__ rvarY,
        const float* __restrict__ gamma, const float* __restrict__ beta,
        bf16* __restrict__ R) {
    const int o   = threadIdx.x;
    const int pix = blockIdx.x;                    // (b*32+hp)*32+wp
    const int wp  = pix & 31, hp = (pix >> 5) & 31, b = pix >> 10;
    float v[4];
#pragma unroll
    for (int s = 0; s < 4; ++s) {
        float y  = Y[((size_t)(s * 16384 + pix)) * 256 + o];
        float xn = (y - meanY[s * 256 + o]) * rvarY[s * 256 + o] * gamma[o] + beta[o];
        v[s] = fmaxf(xn, 0.f);
    }
    float Ad = v[0], Hd = v[1], Vd = v[2], Dd = v[3];
    float lo0 = (Ad + Vd) * SQ, lo1 = (Ad - Vd) * SQ;
    float hi0 = (Hd + Dd) * SQ, hi1 = (Hd - Dd) * SQ;
    bf16* outp = R + (((size_t)(b * 64 + hp * 2) * 64) + wp * 2) * 512 + o;
    outp[0]              = (bf16)((lo0 + hi0) * SQ);
    outp[512]            = (bf16)((lo1 + hi1) * SQ);
    outp[64 * 512]       = (bf16)((lo0 - hi0) * SQ);
    outp[64 * 512 + 512] = (bf16)((lo1 - hi1) * SQ);
}

// ---------------- branch 1: level-3 Haar round trip, 8x8 tile per thread ----------
template <int N>
__device__ __forceinline__ void haar_fwd(float a[8][8]) {
    float u[8][8];
#pragma unroll
    for (int i = 0; i < N / 2; ++i)
#pragma unroll
        for (int j = 0; j < N / 2; ++j) {
            float p00 = a[2*i][2*j], p01 = a[2*i][2*j+1];
            float p10 = a[2*i+1][2*j], p11 = a[2*i+1][2*j+1];
            float lo0 = (p00 + p10) * SQ, lo1 = (p01 + p11) * SQ;
            float hi0 = (p00 - p10) * SQ, hi1 = (p01 - p11) * SQ;
            u[i][j]             = (lo0 + lo1) * SQ;   // A
            u[i][N/2 + j]       = (lo0 - lo1) * SQ;   // V
            u[N/2 + i][j]       = (hi0 + hi1) * SQ;   // H
            u[N/2 + i][N/2 + j] = (hi0 - hi1) * SQ;   // D
        }
#pragma unroll
    for (int i = 0; i < N; ++i)
#pragma unroll
        for (int j = 0; j < N; ++j) a[i][j] = u[i][j];
}

template <int N>
__device__ __forceinline__ void haar_inv(float a[8][8]) {
    float u[8][8];
#pragma unroll
    for (int i = 0; i < N / 2; ++i)
#pragma unroll
        for (int j = 0; j < N / 2; ++j) {
            float Ad = a[i][j], Vd = a[i][N/2 + j];
            float Hd = a[N/2 + i][j], Dd = a[N/2 + i][N/2 + j];
            float lo0 = (Ad + Vd) * SQ, lo1 = (Ad - Vd) * SQ;
            float hi0 = (Hd + Dd) * SQ, hi1 = (Hd - Dd) * SQ;
            u[2*i][2*j]     = (lo0 + hi0) * SQ;
            u[2*i][2*j+1]   = (lo1 + hi1) * SQ;
            u[2*i+1][2*j]   = (lo0 - hi0) * SQ;
            u[2*i+1][2*j+1] = (lo1 - hi1) * SQ;
        }
#pragma unroll
    for (int i = 0; i < N; ++i)
#pragma unroll
        for (int j = 0; j < N; ++j) a[i][j] = u[i][j];
}

__global__ __launch_bounds__(256) void k_branch1(const float* __restrict__ x,
                                                 bf16* __restrict__ R) {
    int tid = blockIdx.x * 256 + threadIdx.x;      // c fastest for coalesced writes
    const int c = tid & 255;
    const int t = (tid >> 8) & 63;                 // 8x8 tile id within image
    const int b = tid >> 14;
    const int ty = t >> 3, tx = t & 7;
    const float* xp = x + (((size_t)(b * 256 + c) * 64 + ty * 8) * 64 + tx * 8);
    float a[8][8];
#pragma unroll
    for (int i = 0; i < 8; ++i)
#pragma unroll
        for (int j = 0; j < 8; ++j) a[i][j] = xp[i * 64 + j];
    haar_fwd<8>(a); haar_fwd<4>(a); haar_fwd<2>(a);
    haar_inv<2>(a); haar_inv<4>(a); haar_inv<8>(a);
    bf16* outp = R + (((size_t)(b * 64 + ty * 8) * 64) + tx * 8) * 512 + 256 + c;
#pragma unroll
    for (int i = 0; i < 8; ++i)
#pragma unroll
        for (int j = 0; j < 8; ++j) outp[((size_t)i * 64 + j) * 512] = (bf16)a[i][j];
}

// ---- 3x3 conv as implicit WMMA GEMM (M=65536,N=256,K=9*512) ----------------------
// Block: 8 waves x (16M x 64N) per wave; per (tap, k-chunk) weight panel in LDS (32 KiB).
__global__ __launch_bounds__(256) void gemm_conv(const bf16* __restrict__ Rm,
        const bf16* __restrict__ Ww, const float* __restrict__ bias,
        float* __restrict__ Z) {
    __shared__ __align__(16) bf16 sW[64 * 256];
    const int nBase  = blockIdx.y * 64;
    const int wave   = threadIdx.x >> 5;
    const int lane   = threadIdx.x & 31;
    const int laneHi = lane >> 4;
    const int row    = lane & 15;
    const int mBase  = (blockIdx.x * 8 + wave) * 16;   // ((b*64+h)*64 + w0)
    const int w0 = mBase & 63;
    const int h  = (mBase >> 6) & 63;
    const int b  = mBase >> 12;
    v8f acc[4] = {};
    for (int tap = 0; tap < 9; ++tap) {
        const int dy = tap / 3 - 1, dx = tap % 3 - 1;
        const int hh = h + dy;
        const int ww = w0 + row + dx;
        const bool ok = (hh >= 0) && (hh < 64) && (ww >= 0) && (ww < 64);
        const bf16* arow = Rm + ((size_t)(b * 64 + hh) * 64 + ww) * 512;
        for (int kc = 0; kc < 512; kc += 256) {
            __syncthreads();
            {   // cooperative LDS fill: 64n x 256k bf16 for this (tap, k-chunk)
                int t = threadIdx.x;
                int n = t >> 2;             // 0..63
                int k = (t & 3) * 64;       // 64 bf16 each
                const bf16* src = Ww + ((size_t)(nBase + n) * 9 + tap) * 512 + kc + k;
                bf16* dst = &sW[n * 256 + k];
#pragma unroll
                for (int q = 0; q < 8; ++q)
                    *(v8bf*)(dst + q * 8) = *(const v8bf*)(src + q * 8);
            }
            __syncthreads();
#pragma unroll
            for (int k0 = 0; k0 < 256; k0 += 32) {
                v8bf a0 = {}, a1 = {};
                if (ok) {
                    a0 = *(const v8bf*)(arow + kc + k0 + laneHi * 8);
                    a1 = *(const v8bf*)(arow + kc + k0 + 16 + laneHi * 8);
                }
                v16bf av = CAT16(a0, a1);
#pragma unroll
                for (int nt = 0; nt < 4; ++nt) {
                    const bf16* brow = &sW[(nt * 16 + row) * 256 + k0];
                    v8bf b0 = *(const v8bf*)(brow + laneHi * 8);
                    v8bf b1 = *(const v8bf*)(brow + 16 + laneHi * 8);
                    v16bf bv = CAT16(b0, b1);
                    acc[nt] = __builtin_amdgcn_wmma_f32_16x16x32_bf16(false, av, false, bv,
                                                                      (short)0, acc[nt],
                                                                      false, false);
                }
            }
        }
    }
#pragma unroll
    for (int nt = 0; nt < 4; ++nt) {
        const float bv = bias[nBase + nt * 16 + row];
        float* zout = Z + (size_t)(mBase + laneHi * 8) * 256 + nBase + nt * 16 + row;
#pragma unroll
        for (int i = 0; i < 8; ++i) zout[(size_t)i * 256] = acc[nt][i] + bv;
    }
}

// ---------------- final BN stats + apply ----------------
__global__ __launch_bounds__(256) void k_statsZ(const float* __restrict__ Z,
                                                float* __restrict__ meanZ,
                                                float* __restrict__ rvarZ) {
    __shared__ float ssum[256], ssq[256];
    const int o = blockIdx.x;
    float sum = 0.f, sq = 0.f;
    for (int i = threadIdx.x; i < 65536; i += 256) {
        float v = Z[(size_t)i * 256 + o];
        sum += v; sq += v * v;
    }
    ssum[threadIdx.x] = sum; ssq[threadIdx.x] = sq;
    __syncthreads();
    for (int st = 128; st > 0; st >>= 1) {
        if (threadIdx.x < st) {
            ssum[threadIdx.x] += ssum[threadIdx.x + st];
            ssq[threadIdx.x]  += ssq[threadIdx.x + st];
        }
        __syncthreads();
    }
    if (threadIdx.x == 0) {
        float m   = ssum[0] * (1.0f / 65536.f);
        float var = ssq[0] * (1.0f / 65536.f) - m * m;
        meanZ[o] = m;
        rvarZ[o] = rsqrtf(var + 1e-5f);
    }
}

__global__ __launch_bounds__(256) void k_bn_out(const float* __restrict__ Z,
        const float* __restrict__ meanZ, const float* __restrict__ rvarZ,
        const float* __restrict__ gamma, const float* __restrict__ beta,
        float* __restrict__ out) {
    const int o = blockIdx.y;
    const int w = threadIdx.x & 63;
    const int h = (blockIdx.x & 15) * 4 + (threadIdx.x >> 6);
    const int b = blockIdx.x >> 4;
    const size_t m = (size_t)(b * 64 + h) * 64 + w;
    float v  = Z[m * 256 + o];
    float xn = (v - meanZ[o]) * rvarZ[o] * gamma[o] + beta[o];
    out[((size_t)(b * 256 + o) * 64 + h) * 64 + w] = fmaxf(xn, 0.f);
}

extern "C" void kernel_launch(void* const* d_in, const int* in_sizes, int n_in,
                              void* d_out, int out_size, void* d_ws, size_t ws_size,
                              hipStream_t stream) {
    (void)in_sizes; (void)n_in; (void)out_size; (void)ws_size;
    const float* x       = (const float*)d_in[0];
    const float* conv_w  = (const float*)d_in[1];
    const float* bn_g    = (const float*)d_in[2];
    const float* bn_b    = (const float*)d_in[3];
    const float* convW_w = (const float*)d_in[4];
    const float* convW_b = (const float*)d_in[5];
    const float* bnW_g   = (const float*)d_in[6];
    const float* bnW_b   = (const float*)d_in[7];
    float* out = (float*)d_out;

    char* ws = (char*)d_ws;
    bf16*  A_sub = (bf16*)(ws);                              // 32 MiB
    float* Y     = (float*)(ws + ((size_t)32  << 20));       // 64 MiB
    bf16*  R     = (bf16*)(ws + ((size_t)96  << 20));        // 64 MiB
    float* Z     = (float*)(ws + ((size_t)32  << 20));       // reuse Y region
    bf16*  wB    = (bf16*)(ws + ((size_t)160 << 20));        // 128 KiB
    bf16*  wW    = (bf16*)(ws + ((size_t)161 << 20));        // 2.25 MiB
    float* meanY = (float*)(ws + ((size_t)164 << 20));
    float* rvarY = meanY + 1024;
    float* meanZ = rvarY + 1024;
    float* rvarZ = meanZ + 256;

    k_cvt_w1<<<256, 256, 0, stream>>>(conv_w, wB);
    k_cvt_w2<<<4608, 256, 0, stream>>>(convW_w, wW);
    k_dwt_sub<<<16384, 256, 0, stream>>>(x, A_sub);
    gemm_sub<<<dim3(512, 4), 256, 0, stream>>>(A_sub, wB, Y);
    k_statsY<<<1024, 256, 0, stream>>>(Y, meanY, rvarY);
    k_bn_idwt<<<16384, 256, 0, stream>>>(Y, meanY, rvarY, bn_g, bn_b, R);
    k_branch1<<<1024, 256, 0, stream>>>(x, R);
    gemm_conv<<<dim3(512, 4), 256, 0, stream>>>(R, wW, convW_b, Z);
    k_statsZ<<<256, 256, 0, stream>>>(Z, meanZ, rvarZ);
    k_bn_out<<<dim3(256, 256), 256, 0, stream>>>(Z, meanZ, rvarZ, bnW_g, bnW_b, out);
}
